// Q_Mlp_57432302682876
// MI455X (gfx1250) — compile-verified, matrix operand order checked by reading, not run on hardware
//
#include <hip/hip_runtime.h>
#include <hip/hip_bf16.h>

// ---------------------------------------------------------------------------
// Quantized MLP (QAT int8-style), MI455X / gfx1250, wave32 WMMA f16 path.
//   B=64 T=196 D=768 H=3072, M = B*T = 12544
//   GEMM1: h   = x  @ w1_int^T + b1_int          (M x 768  @ 768x3072)
//   GEMM2: out = relu(h @ w2_int^T + b2_int)     (M x 3072 @ 3072x768)
// All matmul operands are integer-valued -> exact in f16 (GEMM1) or exact
// after a 256*hi+lo f16 split (GEMM2 activations).
// WG = 256 threads = 8 waves (4M x 2N), macro-tile 128x128, wave tile 32x64.
// ---------------------------------------------------------------------------

typedef __attribute__((ext_vector_type(16))) _Float16 v16h;
typedef __attribute__((ext_vector_type(8)))  _Float16 v8h;
typedef __attribute__((ext_vector_type(8)))  float    v8f;

#define M_ROWS 12544
#define D_DIM  768
#define H_DIM  3072

// ---------------------------------------------------------------------------
// WMMA helper: D = A(16x32 f16) * B(32x16 f16) + C(16x16 f32)
// ---------------------------------------------------------------------------
__device__ __forceinline__ v8f wmma_f16(v16h a, v16h b, v8f c) {
  return __builtin_amdgcn_wmma_f32_16x16x32_f16(
      /*neg_a=*/false, a, /*neg_b=*/false, b,
      /*c_mod=*/(short)0, c, /*reuse_a=*/false, /*reuse_b=*/false);
}

// ---------------------------------------------------------------------------
// Fragment loader. Per CDNA5 ISA (CDNA5 ISA, 16-bit A 16x32 layout):
//   lanes 0-15 : row = lane,    elems[0..7] = K 0..7,  elems[8..15] = K 16..23
//   lanes 16-31: row = lane-16, elems[0..7] = K 8..15, elems[8..15] = K 24..31
// B (32x16, K x N) mirrors this with N as the per-lane "row", so for our
// row-major [N][K] weight arrays A- and B-fragments use identical addressing.
// `base` points at element (row0, k0) of the 16x32 tile; ld = row stride.
// ---------------------------------------------------------------------------
__device__ __forceinline__ v16h frag_f16(const _Float16* __restrict__ base, int ld) {
  const int lane = threadIdx.x & 31;
  const int r = lane & 15, hs = lane >> 4;
  const _Float16* q = base + (size_t)r * ld + hs * 8;
  v8h lo = *reinterpret_cast<const v8h*>(q);
  v8h hi = *reinterpret_cast<const v8h*>(q + 16);
  v16h f;
#pragma unroll
  for (int i = 0; i < 8; ++i) { f[i] = lo[i]; f[i + 8] = hi[i]; }
  return f;
}

// ---------------------------------------------------------------------------
// Pre-pass kernels
// ---------------------------------------------------------------------------
__global__ void init_kernel(unsigned* amax) {
  if (threadIdx.x < 2) amax[threadIdx.x] = 0u;
}

__global__ __launch_bounds__(256) void absmax_kernel(const float* __restrict__ w,
                                                     int n, unsigned* out) {
  __shared__ float red[256];
  float m = 0.f;
  for (int i = blockIdx.x * 256 + threadIdx.x; i < n; i += gridDim.x * 256)
    m = fmaxf(m, fabsf(w[i]));
  red[threadIdx.x] = m;
  __syncthreads();
#pragma unroll
  for (int s = 128; s > 0; s >>= 1) {
    if ((int)threadIdx.x < s)
      red[threadIdx.x] = fmaxf(red[threadIdx.x], red[threadIdx.x + s]);
    __syncthreads();
  }
  if (threadIdx.x == 0) atomicMax(out, __float_as_uint(red[0]));
}

// x values are small integers -> f32->f16 conversion is exact.
__global__ __launch_bounds__(256) void cvt_x_kernel(const float* __restrict__ x,
                                                    _Float16* __restrict__ x16, int n) {
  int i = blockIdx.x * 256 + threadIdx.x;
  if (i < n) x16[i] = (_Float16)x[i];
}

// w_int = clip(rint(w / w_s), +-127), stored exactly as f16
__global__ __launch_bounds__(256) void quant_w_kernel(const float* __restrict__ w,
                                                      _Float16* __restrict__ wq,
                                                      int n, const unsigned* amax) {
  const float ws = __uint_as_float(*amax) * (1.0f / 127.0f);
  int i = blockIdx.x * 256 + threadIdx.x;
  if (i < n) {
    float v = rintf(w[i] / ws);                 // rintf: round-half-even == jnp.round
    v = fminf(fmaxf(v, -127.0f), 127.0f);
    wq[i] = (_Float16)v;
  }
}

// b_int = clip(rint(b / (w_s * prev_scale)), +-(2^31-1)), stored as f32
__global__ __launch_bounds__(256) void quant_b_kernel(const float* __restrict__ b,
                                                      float* __restrict__ bq, int n,
                                                      const unsigned* amax_w,
                                                      const float* __restrict__ a_s,
                                                      const unsigned* amax_prev) {
  const float ws = __uint_as_float(*amax_w) * (1.0f / 127.0f);
  float pre = a_s[0];
  if (amax_prev) pre *= __uint_as_float(*amax_prev) * (1.0f / 127.0f);
  const float denom = ws * pre;
  int i = blockIdx.x * 256 + threadIdx.x;
  if (i < n) {
    float v = rintf(b[i] / denom);
    v = fminf(fmaxf(v, -2147483647.0f), 2147483647.0f);
    bq[i] = v;
  }
}

__global__ void write_scale_kernel(float* dst, const float* __restrict__ a_s,
                                   const unsigned* amax) {
  const float ws1 = __uint_as_float(amax[0]) * (1.0f / 127.0f);
  const float ws2 = __uint_as_float(amax[1]) * (1.0f / 127.0f);
  *dst = a_s[0] * ws1 * ws2;
}

// ---------------------------------------------------------------------------
// GEMM1: h[M,3072] = x16[M,768] @ w1q[3072,768]^T + b1q
// Epilogue: exact split h = 256*h_hi + h_lo  (both f16-exact for |h| < 2^18)
// Grid (3072/128, 12544/128) = (24,98). Wave tile 32(M) x 64(N):
// 2 A-frags + 4 B-frags -> 8 WMMAs per K-step.
// ---------------------------------------------------------------------------
__global__ __launch_bounds__(256) void gemm1_kernel(
    const _Float16* __restrict__ x16, const _Float16* __restrict__ w1q,
    const float* __restrict__ b1q, _Float16* __restrict__ h_hi,
    _Float16* __restrict__ h_lo) {
  const int wave = threadIdx.x >> 5;
  const int m0 = blockIdx.y * 128 + (wave >> 1) * 32;
  const int n0 = blockIdx.x * 128 + (wave & 1) * 64;

  v8f acc[2][4];
#pragma unroll
  for (int i = 0; i < 2; ++i)
#pragma unroll
    for (int j = 0; j < 4; ++j) acc[i][j] = v8f{};

  const _Float16* xa = x16 + (size_t)m0 * D_DIM;
  const _Float16* wb = w1q + (size_t)n0 * D_DIM;

  for (int k = 0; k < D_DIM; k += 32) {
    v16h a0 = frag_f16(xa + k, D_DIM);
    v16h a1 = frag_f16(xa + (size_t)16 * D_DIM + k, D_DIM);
    v16h b[4];
#pragma unroll
    for (int j = 0; j < 4; ++j)
      b[j] = frag_f16(wb + (size_t)(16 * j) * D_DIM + k, D_DIM);
#pragma unroll
    for (int j = 0; j < 4; ++j) {
      acc[0][j] = wmma_f16(a0, b[j], acc[0][j]);
      acc[1][j] = wmma_f16(a1, b[j], acc[1][j]);
    }
  }

  const int lane = threadIdx.x & 31;
  const int col = lane & 15, rb = (lane >> 4) * 8;
#pragma unroll
  for (int ti = 0; ti < 2; ++ti) {
#pragma unroll
    for (int tj = 0; tj < 4; ++tj) {
      const int m = m0 + ti * 16 + rb;
      const int n = n0 + tj * 16 + col;
      const float bias = b1q[n];
#pragma unroll
      for (int i = 0; i < 8; ++i) {
        float v = acc[ti][tj][i] + bias;             // integer-valued, |v| << 2^18
        _Float16 vh = (_Float16)(v * (1.0f / 256.0f));
        float rem = v - 256.0f * (float)vh;          // integer, |rem| <= 128
        size_t idx = (size_t)(m + i) * H_DIM + n;
        h_hi[idx] = vh;
        h_lo[idx] = (_Float16)rem;
      }
    }
  }
}

// ---------------------------------------------------------------------------
// GEMM2: out[M,768] = relu( (256*h_hi + h_lo)[M,3072] @ w2q[768,3072]^T + b2q )
// Grid (768/128, 12544/128) = (6,98). Wave tile 32(M) x 64(N):
// 4 A-frags (hi/lo x 2) + 4 B-frags -> 16 WMMAs per K-step.
// ---------------------------------------------------------------------------
__global__ __launch_bounds__(256) void gemm2_kernel(
    const _Float16* __restrict__ h_hi, const _Float16* __restrict__ h_lo,
    const _Float16* __restrict__ w2q, const float* __restrict__ b2q,
    float* __restrict__ out) {
  const int wave = threadIdx.x >> 5;
  const int m0 = blockIdx.y * 128 + (wave >> 1) * 32;
  const int n0 = blockIdx.x * 128 + (wave & 1) * 64;

  v8f accH[2][4], accL[2][4];
#pragma unroll
  for (int i = 0; i < 2; ++i)
#pragma unroll
    for (int j = 0; j < 4; ++j) { accH[i][j] = v8f{}; accL[i][j] = v8f{}; }

  const size_t arow = (size_t)m0 * H_DIM;
  const _Float16* wb = w2q + (size_t)n0 * H_DIM;

  for (int k = 0; k < H_DIM; k += 32) {
    v16h a0h = frag_f16(h_hi + arow + k, H_DIM);
    v16h a1h = frag_f16(h_hi + arow + (size_t)16 * H_DIM + k, H_DIM);
    v16h a0l = frag_f16(h_lo + arow + k, H_DIM);
    v16h a1l = frag_f16(h_lo + arow + (size_t)16 * H_DIM + k, H_DIM);
    v16h b[4];
#pragma unroll
    for (int j = 0; j < 4; ++j)
      b[j] = frag_f16(wb + (size_t)(16 * j) * H_DIM + k, H_DIM);
#pragma unroll
    for (int j = 0; j < 4; ++j) {
      accH[0][j] = wmma_f16(a0h, b[j], accH[0][j]);
      accH[1][j] = wmma_f16(a1h, b[j], accH[1][j]);
      accL[0][j] = wmma_f16(a0l, b[j], accL[0][j]);
      accL[1][j] = wmma_f16(a1l, b[j], accL[1][j]);
    }
  }

  const int lane = threadIdx.x & 31;
  const int col = lane & 15, rb = (lane >> 4) * 8;
#pragma unroll
  for (int ti = 0; ti < 2; ++ti) {
#pragma unroll
    for (int tj = 0; tj < 4; ++tj) {
      const int m = m0 + ti * 16 + rb;
      const int n = n0 + tj * 16 + col;
      const float bias = b2q[n];
#pragma unroll
      for (int i = 0; i < 8; ++i) {
        float v = 256.0f * accH[ti][tj][i] + accL[ti][tj][i] + bias;
        out[(size_t)(m + i) * D_DIM + n] = fmaxf(v, 0.0f);
      }
    }
  }
}

// ---------------------------------------------------------------------------
// Launch
// ---------------------------------------------------------------------------
extern "C" void kernel_launch(void* const* d_in, const int* in_sizes, int n_in,
                              void* d_out, int out_size, void* d_ws, size_t ws_size,
                              hipStream_t stream) {
  (void)in_sizes; (void)n_in; (void)ws_size;
  const float* x   = (const float*)d_in[0];   // [64,196,768]
  const float* w1  = (const float*)d_in[1];   // [3072,768]
  const float* b1  = (const float*)d_in[2];   // [3072]
  const float* w2  = (const float*)d_in[3];   // [768,3072]
  const float* b2  = (const float*)d_in[4];   // [768]
  const float* a_s = (const float*)d_in[5];   // [1]
  float* out = (float*)d_out;                 // [64,196,768] then scalar s

  // Workspace carve-out (256B aligned slabs)
  char* ws = (char*)d_ws;
  size_t off = 0;
  auto carve = [&](size_t bytes) -> void* {
    void* p = ws + off;
    off += (bytes + 255) & ~(size_t)255;
    return p;
  };
  unsigned* amax  = (unsigned*)carve(2 * sizeof(unsigned));            // [w1,w2]
  _Float16* x16   = (_Float16*)carve((size_t)M_ROWS * D_DIM * 2);
  _Float16* w1q   = (_Float16*)carve((size_t)H_DIM * D_DIM * 2);
  _Float16* w2q   = (_Float16*)carve((size_t)D_DIM * H_DIM * 2);
  float*    b1q   = (float*)   carve((size_t)H_DIM * 4);
  float*    b2q   = (float*)   carve((size_t)D_DIM * 4);
  _Float16* h_hi  = (_Float16*)carve((size_t)M_ROWS * H_DIM * 2);
  _Float16* h_lo  = (_Float16*)carve((size_t)M_ROWS * H_DIM * 2);

  const int NW = H_DIM * D_DIM;    // 2359296 weights per matrix
  const int NX = M_ROWS * D_DIM;   // 9633792 activations

  init_kernel<<<1, 32, 0, stream>>>(amax);
  absmax_kernel<<<512, 256, 0, stream>>>(w1, NW, amax + 0);
  absmax_kernel<<<512, 256, 0, stream>>>(w2, NW, amax + 1);
  cvt_x_kernel<<<(NX + 255) / 256, 256, 0, stream>>>(x, x16, NX);
  quant_w_kernel<<<(NW + 255) / 256, 256, 0, stream>>>(w1, w1q, NW, amax + 0);
  quant_w_kernel<<<(NW + 255) / 256, 256, 0, stream>>>(w2, w2q, NW, amax + 1);
  quant_b_kernel<<<(H_DIM + 255) / 256, 256, 0, stream>>>(b1, b1q, H_DIM, amax + 0, a_s, nullptr);
  quant_b_kernel<<<(D_DIM + 255) / 256, 256, 0, stream>>>(b2, b2q, D_DIM, amax + 1, a_s, amax + 0);
  write_scale_kernel<<<1, 1, 0, stream>>>(out + (out_size - 1), a_s, amax);

  gemm1_kernel<<<dim3(H_DIM / 128, M_ROWS / 128), 256, 0, stream>>>(x16, w1q, b1q, h_hi, h_lo);
  gemm2_kernel<<<dim3(D_DIM / 128, M_ROWS / 128), 256, 0, stream>>>(h_hi, h_lo, w2q, b2q, out);
}
